// CollaborativeEncoder_57071525429473
// MI455X (gfx1250) — matrix operand order repaired and checked by reading, not compiled, not run on hardware
//
#include <hip/hip_runtime.h>

#define N_NODES 100000
#define DIM 64
#define EPS 1e-5f

typedef __attribute__((ext_vector_type(16))) __bf16 v16bf;
typedef __attribute__((ext_vector_type(8)))  float  v8f;

// native hardware f32 -> bf16 convert (RNE on CDNA5)
__device__ __forceinline__ __bf16 f2bf(float f) { return (__bf16)f; }

// Y[m,n] = sum_k X[m,k]*W[k,n] + bias[n], plus per-column sum / sumsq for BN.
// Safe when X == Y (in-place): each block only reads/writes rows m0..m0+15,
// and all global loads complete (s_wait_loadcnt 0) before the block barrier
// that precedes any store.
__global__ __launch_bounds__(128) void gemm_bias_stats(
    const float* X, const float* __restrict__ W,
    const float* __restrict__ bias, float* Y,
    float* __restrict__ stats /* [0..63]=sum [64..127]=sumsq */) {
  const int lane = threadIdx.x & 31;
  const int wave = threadIdx.x >> 5;
  const int m0   = blockIdx.x * 16;
  const int n0   = wave * 16;
  const int half = (lane & 16) ? 1 : 0;      // upper half-wave
  const int mrow = m0 + (lane & 15);
  const int ncol = n0 + (lane & 15);

  // A fragments (16x32 bf16, two K-steps covering K=0..63)
  v16bf a0, a1;
  const float* xrow = X + mrow * DIM;
#pragma unroll
  for (int j = 0; j < 16; ++j) {
    int off = j + half * 16;                 // K index within 32-block
    a0[j] = f2bf(xrow[off]);
    a1[j] = f2bf(xrow[off + 32]);
  }
  // B fragments (32x16 bf16)
  v16bf b0, b1;
  const int kb = half * 16;
#pragma unroll
  for (int j = 0; j < 16; ++j) {
    b0[j] = f2bf(W[(kb + j)      * DIM + ncol]);
    b1[j] = f2bf(W[(kb + j + 32) * DIM + ncol]);
  }

  v8f c = {0.f, 0.f, 0.f, 0.f, 0.f, 0.f, 0.f, 0.f};
  c = __builtin_amdgcn_wmma_f32_16x16x32_bf16(false, a0, false, b0, (short)0, c, false, false);
  c = __builtin_amdgcn_wmma_f32_16x16x32_bf16(false, a1, false, b1, (short)0, c, false, false);

  const float bv = bias[ncol];
  float y[8];
  float s = 0.f, s2 = 0.f;
#pragma unroll
  for (int r = 0; r < 8; ++r) {
    y[r] = c[r] + bv;
    s  += y[r];
    s2 += y[r] * y[r];
  }

  // all reads of X are consumed; make completion explicit, then barrier so
  // no wave's store can race another wave's A-tile load (in-place case).
  asm volatile("s_wait_loadcnt 0x0" ::: "memory");
  __syncthreads();

#pragma unroll
  for (int r = 0; r < 8; ++r)
    Y[(m0 + half * 8 + r) * DIM + ncol] = y[r];

  // lanes l and l+16 hold the same column: pair-reduce, then 16 atomics/wave
  s  += __shfl_down(s, 16, 32);
  s2 += __shfl_down(s2, 16, 32);
  if (lane < 16) {
    atomicAdd(&stats[ncol],      s);
    atomicAdd(&stats[64 + ncol], s2);
  }
}

__global__ void bn_finalize(float* stats, const float* __restrict__ gamma,
                            const float* __restrict__ beta) {
  int n = threadIdx.x;
  if (n < DIM) {
    float inv  = 1.0f / (float)N_NODES;
    float mean = stats[n] * inv;
    float var  = stats[64 + n] * inv - mean * mean;
    float rs   = rsqrtf(var + EPS);
    float sc   = rs * gamma[n];
    stats[128 + n] = sc;                    // scale
    stats[192 + n] = beta[n] - mean * sc;   // shift
  }
}

// vectorized BN apply + ReLU, b128 traffic; grid covers exactly N*DIM/4
__global__ __launch_bounds__(256) void bn_relu4(float4* __restrict__ Y,
                                                const float* __restrict__ stats) {
  int i = blockIdx.x * 256 + threadIdx.x;
  int n = (i * 4) & (DIM - 1);              // column of .x (aligned to 4)
  float4 v = Y[i];
  v.x = fmaxf(v.x * stats[128 + n + 0] + stats[192 + n + 0], 0.f);
  v.y = fmaxf(v.y * stats[128 + n + 1] + stats[192 + n + 1], 0.f);
  v.z = fmaxf(v.z * stats[128 + n + 2] + stats[192 + n + 2], 0.f);
  v.w = fmaxf(v.w * stats[128 + n + 3] + stats[192 + n + 3], 0.f);
  Y[i] = v;
}

__global__ __launch_bounds__(256) void zero_f32(float* __restrict__ p, int n) {
  int i = blockIdx.x * 256 + threadIdx.x;
  if (i < n) p[i] = 0.f;
}

__global__ __launch_bounds__(256) void zero_f32x4(float4* __restrict__ p, int n4) {
  int i = blockIdx.x * 256 + threadIdx.x;
  if (i < n4) p[i] = make_float4(0.f, 0.f, 0.f, 0.f);
}

// One wave per edge; lane covers dims {2*lane, 2*lane+1}. Edge scalars are
// wave-uniform (scalarized loads); H-row gather is a coalesced 256B read.
__global__ __launch_bounds__(256) void spmm_scatter(
    const int* __restrict__ rows, const int* __restrict__ cols,
    const float* __restrict__ vals, const float* __restrict__ H,
    float* __restrict__ A, int nedges) {
  int e = blockIdx.x * 8 + (threadIdx.x >> 5);
  if (e >= nedges) return;
  int lane = threadIdx.x & 31;
  int r = rows[e], c = cols[e];
  float v = vals[e];
  const float2 h = *(const float2*)(H + c * DIM + lane * 2);
  float* dst = A + r * DIM + lane * 2;
  atomicAdd(dst,     v * h.x);
  atomicAdd(dst + 1, v * h.y);
}

extern "C" void kernel_launch(void* const* d_in, const int* in_sizes, int n_in,
                              void* d_out, int out_size, void* d_ws, size_t ws_size,
                              hipStream_t stream) {
  const int*   rows  = (const int*)  d_in[0];
  const int*   cols  = (const int*)  d_in[1];
  const float* vals  = (const float*)d_in[2];
  const float* embed = (const float*)d_in[3];
  const float* W0    = (const float*)d_in[4];
  const float* b0    = (const float*)d_in[5];
  const float* g0    = (const float*)d_in[6];
  const float* be0   = (const float*)d_in[7];
  const float* Ws    = (const float*)d_in[8];
  const float* bs    = (const float*)d_in[9];
  const float* gs    = (const float*)d_in[10];
  const float* bes   = (const float*)d_in[11];
  float* out   = (float*)d_out;
  float* stats = (float*)d_ws;              // 256 floats used
  const int E    = in_sizes[0];
  const int slab = N_NODES * DIM;

  // ---- layer 0: H0 = relu(bn(embed @ W0 + b0)) ----
  zero_f32<<<1, 256, 0, stream>>>(stats, 256);
  gemm_bias_stats<<<N_NODES / 16, 128, 0, stream>>>(embed, W0, b0, out, stats);
  bn_finalize<<<1, 64, 0, stream>>>(stats, g0, be0);
  bn_relu4<<<slab / 4 / 256, 256, 0, stream>>>((float4*)out, stats);

  // ---- hops ----
  for (int k = 0; k < 2; ++k) {
    float* Hk = out + k * slab;
    float* Hn = out + (k + 1) * slab;
    zero_f32x4<<<(slab / 4 + 255) / 256, 256, 0, stream>>>((float4*)Hn, slab / 4);
    spmm_scatter<<<(E + 7) / 8, 256, 0, stream>>>(rows, cols, vals, Hk, Hn, E);
    zero_f32<<<1, 256, 0, stream>>>(stats, 256);
    gemm_bias_stats<<<N_NODES / 16, 128, 0, stream>>>(Hn, Ws + k * DIM * DIM,
                                                      bs + k * DIM, Hn, stats);
    bn_finalize<<<1, 64, 0, stream>>>(stats, gs + k * DIM, bes + k * DIM);
    bn_relu4<<<slab / 4 / 256, 256, 0, stream>>>((float4*)Hn, stats);
  }
}
